// G2Conv2d4_len_feature_19902878449860
// MI455X (gfx1250) — matrix operand order repaired and checked by reading, not compiled, not run on hardware
//
#include <hip/hip_runtime.h>
#include <hip/hip_bf16.h>
#include <stdint.h>

// ---------------------------------------------------------------------------
// Shapes (fixed by the reference): x[16,256,64,64] f32, w[256,256,3,3] f32,
// bias[256] f32 -> out[16,256,64,64] f32.
// ---------------------------------------------------------------------------

typedef __bf16 v16bf  __attribute__((ext_vector_type(16)));
typedef __bf16 bf16x8 __attribute__((ext_vector_type(8)));
typedef float  v8f    __attribute__((ext_vector_type(8)));

typedef unsigned int u32x4 __attribute__((ext_vector_type(4)));
typedef int          i32x8 __attribute__((ext_vector_type(8)));
typedef int          i32x4 __attribute__((ext_vector_type(4)));

__device__ __forceinline__ unsigned short f32_to_bf16_rne(float f) {
    union { float f; unsigned u; } c; c.f = f;
    unsigned u = c.u;
    unsigned r = u + 0x7FFFu + ((u >> 16) & 1u);   // round-to-nearest-even
    return (unsigned short)(r >> 16);
}

// ---------------------------------------------------------------------------
// Tensor Data Mover: one descriptor moves a 3D tile (d0 contiguous bf16
// channels, d1 = y rows, d2 = z planes) global->LDS, inserting 4 DWORDs of
// LDS padding every 16 DWORDs (64B) so a 32-channel group lands on an 80B
// stride (bank-conflict-free for ds_load_b128).
// D# packing per CDNA5 ISA 8.3-8.6.  This toolchain's builtin takes 6 args.
// ---------------------------------------------------------------------------
__device__ __forceinline__ void tdm_load_3d(unsigned lds_byte_off, const void* gptr,
                                            int tile_d0, int tile_d1, int tile_d2,
                                            int estride_d1, int estride_d2) {
    unsigned long long ga = (unsigned long long)(uintptr_t)gptr;
    u32x4 g0;
    g0[0] = 1u;                                    // count=1 (valid), user mode
    g0[1] = lds_byte_off;                          // lds_addr
    g0[2] = (unsigned)(ga & 0xFFFFFFFFu);          // global_addr[31:0]
    g0[3] = (unsigned)(ga >> 32) | 0x80000000u;    // global_addr[56:32] | type=2
    i32x8 g1;
    // wg_mask=0 | data_size=2B(code1) | pad_enable | pad_interval=16DW(code3)
    // | pad_amount=4DW(code3)
    g1[0] = (1 << 16) | (1 << 20) | (3 << 22) | (3 << 25);
    g1[1] = (tile_d0 & 0xFFFF) << 16;                              // tensor_dim0 lo
    g1[2] = ((unsigned)tile_d0 >> 16) | ((tile_d1 & 0xFFFF) << 16);// dim0 hi|dim1 lo
    g1[3] = ((unsigned)tile_d1 >> 16) | ((tile_d0 & 0xFFFF) << 16);// dim1 hi|tile_dim0
    g1[4] = (tile_d1 & 0xFFFF) | ((tile_d2 & 0xFFFF) << 16);       // tile_dim1|tile_dim2
    g1[5] = estride_d1;                                            // dim0_stride lo32
    g1[6] = ((unsigned)estride_d2 & 0xFFFF) << 16;                 // dim0_str hi|dim1_str lo
    g1[7] = (unsigned)estride_d2 >> 16;                            // dim1_stride hi
    i32x4 g2;
    g2[0] = tile_d2;                               // tensor_dim2
    g2[1] = 0; g2[2] = 0; g2[3] = 0;
    i32x4 g3 = {0, 0, 0, 0};
    i32x8 g4 = {0, 0, 0, 0, 0, 0, 0, 0};
    __builtin_amdgcn_tensor_load_to_lds(g0, g1, g2, g3, g4, 0);
}

// ---------------------------------------------------------------------------
// Kernel 0: weight reorder  w[O][I][3][3] f32  ->  wbf[tap][O][I] bf16
// ---------------------------------------------------------------------------
__global__ void k_wprep(const float* __restrict__ w, unsigned short* __restrict__ wbf) {
    const int o = blockIdx.x;        // 256
    const int i = threadIdx.x;       // 256
    const float* src = w + ((size_t)o * 256 + i) * 9;
#pragma unroll
    for (int t = 0; t < 9; ++t)
        wbf[((size_t)t * 256 + o) * 256 + i] = f32_to_bf16_rne(src[t]);
}

// ---------------------------------------------------------------------------
// Kernel 1: per-(b,q) channel-energy argmax over 64x64 pixels.
// ---------------------------------------------------------------------------
__global__ void k_argmax(const float* __restrict__ x, int* __restrict__ wsIdx) {
    __shared__ float sv[256];
    __shared__ int   si[256];
    const int id = blockIdx.x;           // b*4 + q
    const int b = id >> 2, q = id & 3;
    const int tid = threadIdx.x;

    float acc[16];
#pragma unroll
    for (int i = 0; i < 16; ++i) acc[i] = 0.f;

    const float* xp = x + (((size_t)b * 256 + q * 64) * 4096);
    for (int c = 0; c < 64; ++c) {
        const float* cp = xp + (size_t)c * 4096;
#pragma unroll
        for (int i = 0; i < 16; ++i) {
            float v = cp[tid + i * 256];
            acc[i] = fmaf(v, v, acc[i]);
        }
    }
    float best = acc[0]; int bidx = tid;
#pragma unroll
    for (int i = 1; i < 16; ++i) {
        int p = tid + i * 256;
        if (acc[i] > best) { best = acc[i]; bidx = p; }
    }
    sv[tid] = best; si[tid] = bidx;
    __syncthreads();
    for (int s = 128; s > 0; s >>= 1) {
        if (tid < s) {
            float v2 = sv[tid + s]; int i2 = si[tid + s];
            if (v2 > sv[tid] || (v2 == sv[tid] && i2 < si[tid])) { sv[tid] = v2; si[tid] = i2; }
        }
        __syncthreads();
    }
    if (tid == 0) wsIdx[id] = si[0];
}

// ---------------------------------------------------------------------------
// Kernel 2a: zero the halo of the padded NHWC pooled buffer [B][66][66][256].
// ---------------------------------------------------------------------------
__global__ void k_zero(unsigned short* __restrict__ pooled) {
    const int b = blockIdx.x;            // 16
    for (int t = threadIdx.x; t < 260 * 64; t += 256) {
        int p = t >> 6, ch = t & 63;     // 64 x 8B chunks cover 256 bf16 channels
        int hh, ww;
        if (p < 66)       { hh = 0;       ww = p;        }
        else if (p < 132) { hh = 65;      ww = p - 66;   }
        else if (p < 196) { hh = p - 131; ww = 0;        }
        else              { hh = p - 195; ww = 65;       }
        size_t off = (((size_t)b * 66 + hh) * 66 + ww) * 256;
        *(unsigned long long*)(pooled + off + ch * 4) = 0ull;
    }
}

// ---------------------------------------------------------------------------
// Kernel 2b: closed-form guided pool + f32->bf16 + NCHW -> padded NHWC.
// ---------------------------------------------------------------------------
__global__ void k_pool(const float* __restrict__ x, const int* __restrict__ wsIdx,
                       unsigned short* __restrict__ pooled) {
    __shared__ unsigned short sT[64 * 66];   // [c_local][w], padded
    const int id = blockIdx.x;               // (b*64 + h)*4 + q
    const int q = id & 3, h = (id >> 2) & 63, b = id >> 8;
    const int tid = threadIdx.x;

    const int lm = wsIdx[b * 4 + q];
    const int r = lm >> 6, cc = lm & 63;
    // _DIRS: q0=(revH,revW)=(T,T) q1=(T,F) q2=(F,T) q3=(F,F)
    const bool hOK = (q <= 1) ? (r >= h) : (r <= h);

#pragma unroll
    for (int k = 0; k < 16; ++k) {
        int e = tid + k * 256;
        int cl = e >> 6, w = e & 63;
        size_t cbase = ((size_t)b * 256 + q * 64 + cl) * 64;
        float v   = x[(cbase + h) * 64 + w];
        float xsq = v * v;
        float lv  = x[(cbase + r) * 64 + cc];
        float vsq = lv * lv;
        bool wOK = ((q & 1) == 0) ? (cc >= w) : (cc <= w);
        float p = (hOK && wOK) ? fmaxf(xsq, vsq) : xsq;
        sT[cl * 66 + w] = f32_to_bf16_rne(p);
    }
    __syncthreads();
    // padded NHWC write: [(b,h+1,w+1)]*256 + q*64 + c, 2 channels per u32
#pragma unroll
    for (int k = 0; k < 8; ++k) {
        int e = tid + k * 256;
        int w = e >> 5, c2 = (e & 31) * 2;
        unsigned lo = sT[c2 * 66 + w];
        unsigned hi = sT[(c2 + 1) * 66 + w];
        size_t off = (((size_t)b * 66 + h + 1) * 66 + (w + 1)) * 256 + q * 64 + c2;
        *(unsigned*)(pooled + off) = lo | (hi << 16);
    }
}

// ---------------------------------------------------------------------------
// Kernel 3: implicit-GEMM 3x3 conv, v_wmma_f32_16x16x32_bf16.
// WG tile: 128 output pixels (2 rows x 64 cols) x 64 output channels.
// 8 waves, each 2Mx2N register block (4 accumulators) -> 2 ds_load_b128/WMMA.
// Tiles staged by the Tensor Data Mover (tensor_load_to_lds / TENSORcnt).
// ---------------------------------------------------------------------------
__global__ void __launch_bounds__(256, 1)
k_conv(const unsigned short* __restrict__ pooled,
       const unsigned short* __restrict__ wbf,
       const float* __restrict__ bias,
       float* __restrict__ out) {
    // sA: [4][66][40] bf16 = 21120 B (padded rows h0..h0+3, cols 0..65, 32ch+pad)
    // sB: [5][64][40] bf16 = 25600 B (tap half, 64 out-ch, 32 in-ch + pad)
    // sOut (epilogue reuse): [128][67] f32 = 34304 B
    __shared__ __align__(16) unsigned char smem[46720];
    unsigned short* sA = (unsigned short*)smem;
    unsigned short* sB = (unsigned short*)(smem + 21120);
    float* sOut = (float*)smem;

    const int tid = threadIdx.x;
    const int b  = blockIdx.x >> 5;
    const int h0 = (blockIdx.x & 31) << 1;     // two output rows h0, h0+1
    const int n0 = blockIdx.y << 6;

    const int wv = tid >> 5, lane = tid & 31;
    const int wg = wv & 3;            // 16-pixel column group
    const int npair = wv >> 2;        // 0/1: which 32-out-channel pair
    const int mrow = lane & 15;
    const bool hiHalf = lane >= 16;
    const int aoff = hiHalf ? 8 : 0;  // A: K = {off..off+7, off+16..off+23}
    const int koff = hiHalf ? 8 : 0;  // B: K = {koff..koff+7, koff+16..koff+23}

    v8f acc00 = {}, acc01 = {}, acc10 = {}, acc11 = {};

    auto do_tap = [&](int tap, int tl) {
        const int kh = tap / 3, kw = tap % 3;
        const __bf16* ap0 = (const __bf16*)(sA + kh * 2640 + (wg * 16 + mrow + kw) * 40);
        union { v16bf v; bf16x8 h[2]; } a0, a1, bb;
        a0.h[0] = *(const bf16x8*)(ap0 + aoff);
        a0.h[1] = *(const bf16x8*)(ap0 + aoff + 16);
        const __bf16* ap1 = ap0 + 2640;                 // next input row
        a1.h[0] = *(const bf16x8*)(ap1 + aoff);
        a1.h[1] = *(const bf16x8*)(ap1 + aoff + 16);

        const __bf16* bp0 = (const __bf16*)(sB + tl * 2560 + (npair * 32 + mrow) * 40 + koff);
        bb.h[0] = *(const bf16x8*)(bp0);
        bb.h[1] = *(const bf16x8*)(bp0 + 16);
        acc00 = __builtin_amdgcn_wmma_f32_16x16x32_bf16(false, a0.v, false, bb.v, (short)0, acc00, false, false);
        acc10 = __builtin_amdgcn_wmma_f32_16x16x32_bf16(false, a1.v, false, bb.v, (short)0, acc10, false, false);

        const __bf16* bp1 = bp0 + 16 * 40;
        bb.h[0] = *(const bf16x8*)(bp1);
        bb.h[1] = *(const bf16x8*)(bp1 + 16);
        acc01 = __builtin_amdgcn_wmma_f32_16x16x32_bf16(false, a0.v, false, bb.v, (short)0, acc01, false, false);
        acc11 = __builtin_amdgcn_wmma_f32_16x16x32_bf16(false, a1.v, false, bb.v, (short)0, acc11, false, false);
    };

    for (int cb = 0; cb < 8; ++cb) {
        const int cbase = cb * 32;

        if (wv == 0) {
            // A tile: 32ch x 66 cols x 4 rows from padded NHWC pooled
            tdm_load_3d(0u, pooled + ((((size_t)b * 66 + h0) * 66) * 256 + cbase),
                        32, 66, 4, 256, 66 * 256);
            // B tile half 0: 32ch x 64 out-ch x taps 0..4 from wbf[tap][O][I]
            tdm_load_3d(21120u, wbf + (((size_t)n0 * 256) + cbase),
                        32, 64, 5, 256, 256 * 256);
            __builtin_amdgcn_s_wait_tensorcnt((short)0);
        }
        __syncthreads();
#pragma unroll
        for (int tl = 0; tl < 5; ++tl) do_tap(tl, tl);
        __syncthreads();

        if (wv == 0) {
            // B tile half 1: taps 5..8
            tdm_load_3d(21120u, wbf + ((((size_t)5 * 256 + n0) * 256) + cbase),
                        32, 64, 4, 256, 256 * 256);
            __builtin_amdgcn_s_wait_tensorcnt((short)0);
        }
        __syncthreads();
#pragma unroll
        for (int tl = 0; tl < 4; ++tl) do_tap(5 + tl, tl);
        __syncthreads();
    }

    // --- epilogue: LDS transpose for coalesced NCHW stores ---
    {
        const int mbase = wg * 16 + (hiHalf ? 8 : 0);
        const int nl = npair * 32 + mrow;
#pragma unroll
        for (int v = 0; v < 8; ++v) {
            sOut[(mbase + v) * 67 + nl]           = acc00[v];
            sOut[(mbase + v) * 67 + nl + 16]      = acc01[v];
            sOut[(mbase + v + 64) * 67 + nl]      = acc10[v];
            sOut[(mbase + v + 64) * 67 + nl + 16] = acc11[v];
        }
        __syncthreads();
        for (int e = tid; e < 8192; e += 256) {
            int n = e >> 7, rw = e & 127, r = rw >> 6, w = rw & 63;
            out[(((size_t)b * 256 + n0 + n) * 64 + h0 + r) * 64 + w] =
                sOut[(r * 64 + w) * 67 + n] + bias[n0 + n];
        }
    }
}

// ---------------------------------------------------------------------------
extern "C" void kernel_launch(void* const* d_in, const int* in_sizes, int n_in,
                              void* d_out, int out_size, void* d_ws, size_t ws_size,
                              hipStream_t stream) {
    const float* x    = (const float*)d_in[0];
    const float* w    = (const float*)d_in[1];
    const float* bias = (const float*)d_in[2];
    float* out = (float*)d_out;

    char* ws = (char*)d_ws;
    int* wsIdx = (int*)ws;                                      //   256 B
    unsigned short* wbf    = (unsigned short*)(ws + 256);       // 9*256*256*2 B
    unsigned short* pooled = (unsigned short*)(ws + 256 + (size_t)9 * 256 * 256 * 2);
                                                                // 16*66*66*256*2 B
    k_wprep <<<256, 256, 0, stream>>>(w, wbf);
    k_argmax<<<64, 256, 0, stream>>>(x, wsIdx);
    k_zero  <<<16, 256, 0, stream>>>(pooled);
    k_pool  <<<16 * 64 * 4, 256, 0, stream>>>(x, wsIdx, pooled);
    k_conv  <<<dim3(16 * 32, 4), 256, 0, stream>>>(pooled, wbf, bias, out);
}
